// Attention_29463475650935
// MI455X (gfx1250) — compile-verified, matrix operand order checked by reading, not run on hardware
//
#include <hip/hip_runtime.h>

typedef __attribute__((ext_vector_type(16))) _Float16 v16h;
typedef __attribute__((ext_vector_type(8)))  _Float16 v8h;
typedef __attribute__((ext_vector_type(8)))  float    v8f;

#define TTOK    4096
#define SEQLEN  1024
#define NHEADS  16
#define HEADD   64
#define HDTOT   1024        // NHEADS*HEADD
#define NSLOTS  16384
#define ATTN_SCALE 0.125f
#define NEG_BIG (-3.0e38f)

#define QSTRIDE 72          // padded half-stride for sQ/sK rows (144B, 16B-aligned)
#define VSTRIDE 40          // padded half-stride for sVT rows (80B, 16B-aligned)

static __device__ __forceinline__ v8f wmma_f16(v16h a, v16h b, v8f c) {
    return __builtin_amdgcn_wmma_f32_16x16x32_f16(false, a, false, b,
                                                  (short)0, c, false, false);
}

// Load a v16h A/B fragment: two 16B runs at (p+off) and (p+off+16) halves.
static __device__ __forceinline__ v16h make_frag(const _Float16* p, int off) {
    v8h lo = *(const v8h*)(p + off);
    v8h hi = *(const v8h*)(p + off + 16);
    v16h f;
#pragma unroll
    for (int i = 0; i < 8; ++i) { f[i] = lo[i]; f[i + 8] = hi[i]; }
    return f;
}

// ---------------- cache copy (old cache -> new cache output) ----------------
__global__ void cache_copy_f4(const float4* __restrict__ src,
                              float4* __restrict__ dst, int n4) {
    int i = blockIdx.x * blockDim.x + threadIdx.x;
    if (i < n4) dst[i] = src[i];
}

// ---------------- slot-mapped scatter of K/V rows into caches ---------------
__global__ void kv_scatter(const float* __restrict__ k,
                           const float* __restrict__ v,
                           const int*   __restrict__ slots,
                           float* __restrict__ kc, float* __restrict__ vc) {
    int tok = blockIdx.x;          // 0..TTOK-1
    int c4  = threadIdx.x;         // 0..255 (256 float4 = 1024 floats/row)
    int slot = slots[tok];
    const float4* ks = (const float4*)(k + (size_t)tok * HDTOT);
    const float4* vs = (const float4*)(v + (size_t)tok * HDTOT);
    float4* kd = (float4*)(kc + (size_t)slot * HDTOT);
    float4* vd = (float4*)(vc + (size_t)slot * HDTOT);
    kd[c4] = ks[c4];
    vd[c4] = vs[c4];
}

// ---------------- flash attention (causal, per (batch, head)) ---------------
__global__ __launch_bounds__(128)
void flash_attn_kernel(const float* __restrict__ q,
                       const float* __restrict__ k,
                       const float* __restrict__ v,
                       float* __restrict__ out) {
    __shared__ __align__(16) _Float16 sQ [64 * QSTRIDE];
    __shared__ __align__(16) _Float16 sK [32 * QSTRIDE];
    __shared__ __align__(16) _Float16 sVT[HEADD * VSTRIDE];

    const int qblk = blockIdx.x;   // 0..15 (64 queries per block)
    const int hh   = blockIdx.y;   // head
    const int bb   = blockIdx.z;   // batch
    const int tid  = threadIdx.x;
    const int wid  = tid >> 5;     // wave 0..3 -> 16 queries each
    const int lane = tid & 31;
    const int col16 = lane & 15;
    const int hsel  = (lane & 16) ? 8 : 0;   // run selector for frag layouts

    const size_t rowbase = (size_t)(bb * SEQLEN) * HDTOT + (size_t)hh * HEADD;
    const float* qg = q + rowbase + (size_t)(qblk * 64) * HDTOT;

    // ---- stage Q tile (64 x 64) fp32 -> f16 ----
    for (int e = tid; e < 64 * HEADD; e += 128) {
        int r = e >> 6, c = e & 63;
        sQ[r * QSTRIDE + c] = (_Float16)qg[(size_t)r * HDTOT + c];
    }
    __syncthreads();

    // ---- per-wave Q B-fragments (lane = query column) ----
    const _Float16* qrow = &sQ[(wid * 16 + col16) * QSTRIDE];
    const v16h qf0 = make_frag(qrow,      hsel);   // d in {0..31}
    const v16h qf1 = make_frag(qrow + 32, hsel);   // d in {32..63}

    const int qw = qblk * 64 + wid * 16;           // wave's first query row
    float m_run = NEG_BIG, l_run = 0.0f;
    v8f acc0 = {}, acc1 = {}, acc2 = {}, acc3 = {};

    const int ntiles = qblk * 2 + 2;               // 32-key tiles (causal)
    for (int t = 0; t < ntiles; ++t) {
        __syncthreads();
        // ---- stage K (32x64) and V^T (64x32) for keys [t*32, t*32+32) ----
        const float* kg = k + rowbase + (size_t)(t * 32) * HDTOT;
        const float* vg = v + rowbase + (size_t)(t * 32) * HDTOT;
        for (int e = tid; e < 32 * HEADD; e += 128) {
            int r = e >> 6, c = e & 63;
            float kv = kg[(size_t)r * HDTOT + c];
            float vv = vg[(size_t)r * HDTOT + c];
            sK [r * QSTRIDE + c] = (_Float16)kv;
            sVT[c * VSTRIDE + r] = (_Float16)vv;   // transposed
        }
        __syncthreads();

        if (t * 32 <= qw + 15) {                   // wave-uniform causal skip
            // ---- S^T tiles: T_s = K_s (A) x Q (B), D: lane=query, vgpr=key
            const _Float16* krow0 = &sK[(col16)      * QSTRIDE];
            const _Float16* krow1 = &sK[(16 + col16) * QSTRIDE];
            v16h ka0 = make_frag(krow0,      hsel);
            v16h ka1 = make_frag(krow0 + 32, hsel);
            v16h kb0 = make_frag(krow1,      hsel);
            v16h kb1 = make_frag(krow1 + 32, hsel);

            v8f t0 = {}, t1 = {};
            t0 = wmma_f16(ka0, qf0, t0);
            t0 = wmma_f16(ka1, qf1, t0);
            t1 = wmma_f16(kb0, qf0, t1);
            t1 = wmma_f16(kb1, qf1, t1);

            // ---- online softmax (per query column) ----
            const int qglob = qw + col16;
            float mloc = NEG_BIG;
#pragma unroll
            for (int r = 0; r < 8; ++r) {
                int key0 = t * 32 + r + hsel;
                float x0 = t0[r] * ATTN_SCALE;
                x0 = (key0 <= qglob) ? x0 : NEG_BIG;
                t0[r] = x0; mloc = fmaxf(mloc, x0);
                int key1 = key0 + 16;
                float x1 = t1[r] * ATTN_SCALE;
                x1 = (key1 <= qglob) ? x1 : NEG_BIG;
                t1[r] = x1; mloc = fmaxf(mloc, x1);
            }
            mloc = fmaxf(mloc, __shfl_xor(mloc, 16, 32));
            float mnew = fmaxf(m_run, mloc);
            float corr = __expf(m_run - mnew);
            float rsum = 0.0f;
#pragma unroll
            for (int r = 0; r < 8; ++r) {
                float e0 = __expf(t0[r] - mnew); t0[r] = e0; rsum += e0;
                float e1 = __expf(t1[r] - mnew); t1[r] = e1; rsum += e1;
            }
            rsum += __shfl_xor(rsum, 16, 32);
            l_run = l_run * corr + rsum;
            m_run = mnew;

            // ---- rescale O accumulators (query = vgpr r + hsel) ----
            float cbr[8];
#pragma unroll
            for (int r = 0; r < 8; ++r) cbr[r] = __shfl(corr, r + hsel, 32);
#pragma unroll
            for (int r = 0; r < 8; ++r) {
                acc0[r] *= cbr[r]; acc1[r] *= cbr[r];
                acc2[r] *= cbr[r]; acc3[r] *= cbr[r];
            }

            // ---- P already in A-fragment layout: pf[h<8]=t0[h], else t1 ----
            v16h pf;
#pragma unroll
            for (int i = 0; i < 8; ++i) {
                pf[i]     = (_Float16)t0[i];
                pf[i + 8] = (_Float16)t1[i];
            }

            // ---- O += P x V  (B-fragments from transposed V in LDS) ----
            v16h vb0 = make_frag(&sVT[(col16)      * VSTRIDE], hsel);
            v16h vb1 = make_frag(&sVT[(16 + col16) * VSTRIDE], hsel);
            v16h vb2 = make_frag(&sVT[(32 + col16) * VSTRIDE], hsel);
            v16h vb3 = make_frag(&sVT[(48 + col16) * VSTRIDE], hsel);
            acc0 = wmma_f16(pf, vb0, acc0);
            acc1 = wmma_f16(pf, vb1, acc1);
            acc2 = wmma_f16(pf, vb2, acc2);
            acc3 = wmma_f16(pf, vb3, acc3);
        }
    }

    // ---- normalize and store O (D layout: lane=dim col, vgpr=query row) ----
    float linv = 1.0f / l_run;
    float nbr[8];
#pragma unroll
    for (int r = 0; r < 8; ++r) nbr[r] = __shfl(linv, r + hsel, 32);

#pragma unroll
    for (int r = 0; r < 8; ++r) {
        int qrowg = qw + r + hsel;
        size_t base = (size_t)(bb * SEQLEN + qrowg) * HDTOT
                    + (size_t)hh * HEADD + col16;
        out[base +  0] = acc0[r] * nbr[r];
        out[base + 16] = acc1[r] * nbr[r];
        out[base + 32] = acc2[r] * nbr[r];
        out[base + 48] = acc3[r] * nbr[r];
    }
}

// ------------------------------- launcher -----------------------------------
extern "C" void kernel_launch(void* const* d_in, const int* in_sizes, int n_in,
                              void* d_out, int out_size, void* d_ws, size_t ws_size,
                              hipStream_t stream) {
    const float* q       = (const float*)d_in[0];
    const float* k       = (const float*)d_in[1];
    const float* v       = (const float*)d_in[2];
    const float* k_cache = (const float*)d_in[3];
    const float* v_cache = (const float*)d_in[4];
    const int*   slots   = (const int*)  d_in[5];
    // d_in[6] = seq_len scalar (fixed at 1024 for this problem shape)

    float* o      = (float*)d_out;
    float* kc_new = o + (size_t)TTOK * HDTOT;
    float* vc_new = kc_new + (size_t)NSLOTS * HDTOT;

    // 1) copy old caches into output cache regions
    const int n4 = (NSLOTS * HDTOT) / 4;           // 4,194,304 float4 each
    cache_copy_f4<<<(n4 + 255) / 256, 256, 0, stream>>>(
        (const float4*)k_cache, (float4*)kc_new, n4);
    cache_copy_f4<<<(n4 + 255) / 256, 256, 0, stream>>>(
        (const float4*)v_cache, (float4*)vc_new, n4);

    // 2) scatter new K/V token rows into the caches
    kv_scatter<<<TTOK, 256, 0, stream>>>(k, v, slots, kc_new, vc_new);

    // 3) causal flash attention
    dim3 grid(SEQLEN / 64, NHEADS, TTOK / SEQLEN);
    flash_attn_kernel<<<grid, 128, 0, stream>>>(q, k, v, o);
}
